// PLMFold_35278861369840
// MI455X (gfx1250) — compile-verified, matrix operand order checked by reading, not run on hardware
//
#include <hip/hip_runtime.h>
#include <math.h>

typedef __bf16 bf16_t;
typedef __attribute__((ext_vector_type(16))) __bf16 v16bf;
typedef __attribute__((ext_vector_type(8)))  __bf16 v8bf;
typedef __attribute__((ext_vector_type(4)))  __bf16 v4bf;
typedef __attribute__((ext_vector_type(8)))  float  v8f;

#define NSEQ 256
#define CMD  384
#define CZD  128

__device__ inline v8f wmma_bf16(v16bf a, v16bf b, v8f c) {
  return __builtin_amdgcn_wmma_f32_16x16x32_bf16(false, a, false, b, (short)0, c, false, false);
}

// load a 16-element bf16 fragment from a 16B-aligned row: elements [lh*8..lh*8+7] and [16+lh*8..+7]
__device__ inline v16bf frag_row(const bf16_t* row, int lh) {
  const v8bf* p = (const v8bf*)row;
  v8bf lo = p[lh];
  v8bf hi = p[2 + lh];
  return __builtin_shufflevector(lo, hi, 0, 1, 2, 3, 4, 5, 6, 7, 8, 9, 10, 11, 12, 13, 14, 15);
}

__device__ inline v4bf cvt4(float4 v) {
  v4bf r;
  r[0] = (bf16_t)v.x; r[1] = (bf16_t)v.y; r[2] = (bf16_t)v.z; r[3] = (bf16_t)v.w;
  return r;
}

// ---------------------------------------------------------------- copy (vectorized, n multiple of 4)
__global__ void copy_kernel(const float* __restrict__ s, float* __restrict__ d, int n4) {
  int i = blockIdx.x * blockDim.x + threadIdx.x;
  if (i < n4) ((float4*)d)[i] = ((const float4*)s)[i];
}

// ---------------------------------------------------------------- layernorm (optionally transposed read of [n,n,C])
__global__ void ln_kernel(const float* __restrict__ x, const float* __restrict__ p,
                          float* __restrict__ out, int C, int transpose, int n) {
  int row = blockIdx.x;
  int srow = row;
  if (transpose) { int i = row / n, j = row % n; srow = j * n + i; }
  const float* xr = x + (size_t)srow * C;
  const float* scale = p;
  const float* bias  = p + C;
  __shared__ float red[2][8];
  float s = 0.f, ss = 0.f;
  for (int c = threadIdx.x * 4; c < C; c += blockDim.x * 4) {
    float4 v = *(const float4*)(xr + c);
    s  += v.x + v.y + v.z + v.w;
    ss += v.x * v.x + v.y * v.y + v.z * v.z + v.w * v.w;
  }
  for (int m = 1; m < 32; m <<= 1) { s += __shfl_xor(s, m, 32); ss += __shfl_xor(ss, m, 32); }
  int wid = threadIdx.x >> 5;
  if ((threadIdx.x & 31) == 0) { red[0][wid] = s; red[1][wid] = ss; }
  __syncthreads();
  int nw = blockDim.x >> 5;
  s = 0.f; ss = 0.f;
  for (int w = 0; w < nw; ++w) { s += red[0][w]; ss += red[1][w]; }
  float mu = s / C;
  float var = ss / C - mu * mu;
  float inv = rsqrtf(var + 1e-5f);
  float* orow = out + (size_t)row * C;
  for (int c = threadIdx.x * 4; c < C; c += blockDim.x * 4) {
    float4 v = *(const float4*)(xr + c);
    float4 sc = *(const float4*)(scale + c);
    float4 bi = *(const float4*)(bias + c);
    float4 o;
    o.x = (v.x - mu) * inv * sc.x + bi.x;
    o.y = (v.y - mu) * inv * sc.y + bi.y;
    o.z = (v.z - mu) * inv * sc.z + bi.z;
    o.w = (v.w - mu) * inv * sc.w + bi.w;
    *(float4*)(orow + c) = o;
  }
}

// ---------------------------------------------------------------- generic WMMA GEMM, double-buffered / software pipelined
// out[r',col] = act(alpha*(A@W) + bias) * aux * rowmask + residual ; r' transposed if store_tr
template <bool FASTW>
__global__ void gemm_kernel(const float* __restrict__ A, const float* __restrict__ W,
                            const float* __restrict__ bias, float* __restrict__ out,
                            const float* __restrict__ aux, const float* __restrict__ rowmask,
                            const float* __restrict__ residual,
                            int K, int Nc, float alpha, int act, int store_tr, int NT) {
  __shared__ bf16_t As[2][64][40];    // row-major over k, padded rows (16B multiple)
  __shared__ bf16_t WsT[2][32][40];   // transposed: [col][k]
  int r0 = blockIdx.x * 64;
  int c0 = blockIdx.y * 32;
  int tid = threadIdx.x;
  int wid = tid >> 5, lane = tid & 31;
  int wr = wid >> 1, wc = wid & 1;
  int m = lane & 15, lh = lane >> 4;
  int aI0 = (tid * 4) >> 5, aK = (tid * 4) & 31;
  int aI1 = aI0 + 32;
  int wK = (tid * 4) >> 5, wF = (tid * 4) & 31;
  const float* Arow0 = A + (size_t)(r0 + aI0) * K + aK;
  const float* Arow1 = A + (size_t)(r0 + aI1) * K + aK;
  const float* Wrow  = W + (size_t)wK * Nc + c0 + wF;

  float4 a0v, a1v, w0v;
  // prologue fetch k0 = 0
  a0v = *(const float4*)Arow0;
  a1v = *(const float4*)Arow1;
  if (FASTW) {
    w0v = *(const float4*)Wrow;
  } else {
    w0v = make_float4(0.f, 0.f, 0.f, 0.f);
    if (c0 + wF + 0 < Nc) w0v.x = Wrow[0];
    if (c0 + wF + 1 < Nc) w0v.y = Wrow[1];
    if (c0 + wF + 2 < Nc) w0v.z = Wrow[2];
    if (c0 + wF + 3 < Nc) w0v.w = Wrow[3];
  }
  *(v4bf*)&As[0][aI0][aK] = cvt4(a0v);
  *(v4bf*)&As[0][aI1][aK] = cvt4(a1v);
  WsT[0][wF + 0][wK] = (bf16_t)w0v.x;
  WsT[0][wF + 1][wK] = (bf16_t)w0v.y;
  WsT[0][wF + 2][wK] = (bf16_t)w0v.z;
  WsT[0][wF + 3][wK] = (bf16_t)w0v.w;
  __syncthreads();

  v8f acc = {};
  int nt = K >> 5;
  for (int t = 0; t < nt; ++t) {
    int nk = (t + 1) * 32;
    if (t + 1 < nt) {   // prefetch next tile into registers (overlaps with WMMA below)
      a0v = *(const float4*)(Arow0 + nk);
      a1v = *(const float4*)(Arow1 + nk);
      if (FASTW) {
        w0v = *(const float4*)(Wrow + (size_t)nk * Nc);
      } else {
        const float* wp = Wrow + (size_t)nk * Nc;
        w0v = make_float4(0.f, 0.f, 0.f, 0.f);
        if (c0 + wF + 0 < Nc) w0v.x = wp[0];
        if (c0 + wF + 1 < Nc) w0v.y = wp[1];
        if (c0 + wF + 2 < Nc) w0v.z = wp[2];
        if (c0 + wF + 3 < Nc) w0v.w = wp[3];
      }
    }
    int cur = t & 1;
    v16bf af  = frag_row(&As[cur][wr * 16 + m][0], lh);
    v16bf bfr = frag_row(&WsT[cur][wc * 16 + m][0], lh);
    acc = wmma_bf16(af, bfr, acc);
    if (t + 1 < nt) {
      int nb = cur ^ 1;
      *(v4bf*)&As[nb][aI0][aK] = cvt4(a0v);
      *(v4bf*)&As[nb][aI1][aK] = cvt4(a1v);
      WsT[nb][wF + 0][wK] = (bf16_t)w0v.x;
      WsT[nb][wF + 1][wK] = (bf16_t)w0v.y;
      WsT[nb][wF + 2][wK] = (bf16_t)w0v.z;
      WsT[nb][wF + 3][wK] = (bf16_t)w0v.w;
    }
    __syncthreads();
  }
  int col = c0 + wc * 16 + m;
  if (FASTW || col < Nc) {
#pragma unroll
    for (int v = 0; v < 8; ++v) {
      int r = r0 + wr * 16 + v + 8 * lh;
      float val = alpha * acc[v];
      if (bias) val += bias[col];
      if (act == 1) val = fmaxf(val, 0.f);
      else if (act == 2) val = 1.f / (1.f + __expf(-val));
      if (aux) val *= aux[(size_t)r * Nc + col];
      if (rowmask) val *= rowmask[r];
      size_t rs = store_tr ? ((size_t)(r % NT) * NT + (r / NT)) : (size_t)r;
      if (residual) val += residual[rs * Nc + col];
      out[rs * Nc + col] = val;
    }
  }
}

// ---------------------------------------------------------------- OPM: PA[d,b,f] += (sum_{c,e} left[b,c]*right[d,e]*W[c,e,f] + ob[f]) / (1e-3 + sm[d]*sm[b])
__global__ void opm_kernel(const float* __restrict__ left, const float* __restrict__ right,
                           const float* __restrict__ W, const float* __restrict__ ob,
                           const float* __restrict__ sm, float* __restrict__ PA) {
  __shared__ float  lefts[64][32];
  __shared__ bf16_t WsT[2][32][40];
  int r0 = blockIdx.x * 64;
  int c0 = blockIdx.y * 32;
  int d = r0 >> 8;
  int tid = threadIdx.x;
  int wid = tid >> 5, lane = tid & 31;
  int wr = wid >> 1, wc = wid & 1;
  int m = lane & 15, lh = lane >> 4;
  // preload left tile (fp32) once
#pragma unroll
  for (int t = 0; t < 2; ++t) {
    int idx = tid * 4 + t * 1024;
    int i = idx >> 5, cc = idx & 31;
    *(float4*)&lefts[i][cc] = *(const float4*)(left + ((r0 & 255) + i) * 32 + cc);
  }
  // per-lane right fragment (fixed across all k-steps)
  float rvf[16];
#pragma unroll
  for (int i = 0; i < 8; ++i) {
    rvf[i]     = right[d * 32 + lh * 8 + i];
    rvf[8 + i] = right[d * 32 + 16 + lh * 8 + i];
  }
  int wE = (tid * 4) >> 5, wF = (tid * 4) & 31;
  const float* Wbase = W + (size_t)wE * CZD + c0 + wF;
  float4 w0v = *(const float4*)Wbase;             // ks = 0
  WsT[0][wF + 0][wE] = (bf16_t)w0v.x;
  WsT[0][wF + 1][wE] = (bf16_t)w0v.y;
  WsT[0][wF + 2][wE] = (bf16_t)w0v.z;
  WsT[0][wF + 3][wE] = (bf16_t)w0v.w;
  __syncthreads();
  v8f acc = {};
  for (int ks = 0; ks < 32; ++ks) {       // k = (c=ks, e=0..31), K = 1024
    if (ks + 1 < 32) w0v = *(const float4*)(Wbase + (size_t)(ks + 1) * 32 * CZD);
    float l = lefts[wr * 16 + m][ks];
    v16bf af;
#pragma unroll
    for (int i = 0; i < 16; ++i) af[i] = (bf16_t)(l * rvf[i]);
    v16bf bfr = frag_row(&WsT[ks & 1][wc * 16 + m][0], lh);
    acc = wmma_bf16(af, bfr, acc);
    if (ks + 1 < 32) {
      int nb = (ks + 1) & 1;
      WsT[nb][wF + 0][wE] = (bf16_t)w0v.x;
      WsT[nb][wF + 1][wE] = (bf16_t)w0v.y;
      WsT[nb][wF + 2][wE] = (bf16_t)w0v.z;
      WsT[nb][wF + 3][wE] = (bf16_t)w0v.w;
    }
    __syncthreads();
  }
  int col = c0 + wc * 16 + m;
#pragma unroll
  for (int v = 0; v < 8; ++v) {
    int r = r0 + wr * 16 + v + 8 * lh;
    int b = r & 255;
    float val = (acc[v] + ob[col]) / (1e-3f + sm[d] * sm[b]);
    PA[(size_t)r * CZD + col] += val;
  }
}

// ---------------------------------------------------------------- pack fp32 [n,n,C] -> bf16 channel-major [C][a][b]
// mode 0: T[c,a,b] = S[a,b,c] ; mode 1: T[c,a,b] = S[b,a,c]
__global__ void pack_kernel(const float* __restrict__ src, bf16_t* __restrict__ dst, int mode) {
  __shared__ bf16_t tile[16][17];
  int a = blockIdx.x, b0 = blockIdx.y * 16, c0 = blockIdx.z * 16;
  int tb = threadIdx.x >> 4, tc = threadIdx.x & 15;
  size_t sidx = (mode == 0) ? ((size_t)a * NSEQ + (b0 + tb)) : ((size_t)(b0 + tb) * NSEQ + a);
  tile[tb][tc] = (bf16_t)src[sidx * CZD + c0 + tc];
  __syncthreads();
  int c = threadIdx.x >> 4, b = threadIdx.x & 15;
  dst[(size_t)(c0 + c) * NSEQ * NSEQ + (size_t)a * NSEQ + b0 + b] = tile[b][c];
}

// ---------------------------------------------------------------- triangle multiply core: per-channel 256x256x256 GEMM
// O[i,j,c] = sum_k At[c,i,k] * Bt[c,j,k]   (both operands row-major over k)
__global__ void trimul_kernel(const bf16_t* __restrict__ At, const bf16_t* __restrict__ Bt,
                              float* __restrict__ O) {
  __shared__ float stg[16][16][8];
  int tile = blockIdx.x;
  int cb = blockIdx.y;
  int wid = threadIdx.x >> 5, lane = threadIdx.x & 31;
  int c = cb * 8 + wid;
  int i0 = (tile >> 4) << 4, j0 = (tile & 15) << 4;
  const bf16_t* A = At + (size_t)c * NSEQ * NSEQ;
  const bf16_t* B = Bt + (size_t)c * NSEQ * NSEQ;
  int m = lane & 15, lh = lane >> 4;
  v8f acc = {};
#pragma unroll
  for (int k0 = 0; k0 < NSEQ; k0 += 32) {
    v16bf af  = frag_row(A + (size_t)(i0 + m) * NSEQ + k0, lh);
    v16bf bfr = frag_row(B + (size_t)(j0 + m) * NSEQ + k0, lh);
    acc = wmma_bf16(af, bfr, acc);
  }
#pragma unroll
  for (int v = 0; v < 8; ++v) stg[v + 8 * lh][m][wid] = acc[v];
  __syncthreads();
  int j = threadIdx.x & 15, i = (threadIdx.x >> 4) & 15;
  float* orow = O + ((size_t)(i0 + i) * NSEQ + (j0 + j)) * CZD + cb * 8;
#pragma unroll
  for (int cc = 0; cc < 8; ++cc) orow[cc] = stg[i][j][cc];
}

// ---------------------------------------------------------------- flash attention with pair bias (WMMA QK^T and P.V)
// Q/K/V: [B*N, H*D] f32 ; NB: [N*N, H] ; mask modes: 0 mask[k], 1 mask[b*N+k], 2 mask[k*N+b]
__global__ void attn_kernel(const float* __restrict__ Q, const float* __restrict__ Kp,
                            const float* __restrict__ Vp, const float* __restrict__ NB,
                            const float* __restrict__ mask, float* __restrict__ WA,
                            int H, int D, int Dp, int maskmode) {
  __shared__ bf16_t qs[4][16][72];   // [wave][q][d]
  __shared__ bf16_t ks[32][72];      // [key][d]
  __shared__ bf16_t vsT[64][40];     // [d][key]
  __shared__ bf16_t ps[4][16][40];   // [wave][q][key32]
  const int Nn = NSEQ;
  int b = blockIdx.z, h = blockIdx.y;
  int wid = threadIdx.x >> 5, lane = threadIdx.x & 31;
  int q0 = blockIdx.x * 64 + wid * 16;
  int nn = lane & 15, lh = lane >> 4;
  int HD = H * D;
  int dsh = (Dp == 64) ? 6 : 5;
  int dmask = Dp - 1;
  // stage Q (per wave)
  for (int idx = lane * 4; idx < 16 * Dp; idx += 128) {
    int qm = idx >> dsh, dd = idx & dmask;
    float4 v = make_float4(0.f, 0.f, 0.f, 0.f);
    if (dd < D) v = *(const float4*)(Q + ((size_t)b * Nn + q0 + qm) * HD + h * D + dd);
    *(v4bf*)&qs[wid][qm][dd] = cvt4(v);
  }
  int nqk = Dp >> 5, nacc = Dp >> 4;
  v16bf qa[2];
  for (int dc = 0; dc < nqk; ++dc) qa[dc] = frag_row(&qs[wid][nn][dc * 32], lh);
  v8f zero = {};
  v8f acc[4];
  for (int f = 0; f < 4; ++f) acc[f] = zero;
  float rmax[8], rsum[8];
#pragma unroll
  for (int v = 0; v < 8; ++v) { rmax[v] = -1e30f; rsum[v] = 0.f; }

  for (int k0 = 0; k0 < Nn; k0 += 32) {
    __syncthreads();
    for (int idx = threadIdx.x * 4; idx < 32 * Dp; idx += 512) {
      int kk = idx >> dsh, dd = idx & dmask;
      float4 kv = make_float4(0.f, 0.f, 0.f, 0.f);
      float4 vv = make_float4(0.f, 0.f, 0.f, 0.f);
      if (dd < D) {
        kv = *(const float4*)(Kp + ((size_t)b * Nn + k0 + kk) * HD + h * D + dd);
        vv = *(const float4*)(Vp + ((size_t)b * Nn + k0 + kk) * HD + h * D + dd);
      }
      *(v4bf*)&ks[kk][dd] = cvt4(kv);
      vsT[dd + 0][kk] = (bf16_t)vv.x;
      vsT[dd + 1][kk] = (bf16_t)vv.y;
      vsT[dd + 2][kk] = (bf16_t)vv.z;
      vsT[dd + 3][kk] = (bf16_t)vv.w;
    }
    __syncthreads();
    v8f sf[2];
    for (int g = 0; g < 2; ++g) {
      v8f s = zero;
      for (int dc = 0; dc < nqk; ++dc) {
        v16bf kb = frag_row(&ks[g * 16 + nn][dc * 32], lh);
        s = wmma_bf16(qa[dc], kb, s);
      }
      sf[g] = s;
    }
#pragma unroll
    for (int g = 0; g < 2; ++g) {
      int key = k0 + g * 16 + nn;
      float mb;
      if (maskmode == 0)      mb = 1e9f * (mask[key] - 1.f);
      else if (maskmode == 1) mb = 1e9f * (mask[(size_t)b * Nn + key] - 1.f);
      else                    mb = 1e9f * (mask[(size_t)key * Nn + b] - 1.f);
#pragma unroll
      for (int v = 0; v < 8; ++v) {
        int q = q0 + v + 8 * lh;
        sf[g][v] += mb + NB[((size_t)q * Nn + key) * H + h];
      }
    }
#pragma unroll
    for (int v = 0; v < 8; ++v) {
      float lm = fmaxf(sf[0][v], sf[1][v]);
      for (int msk = 1; msk <= 8; msk <<= 1) lm = fmaxf(lm, __shfl_xor(lm, msk, 32));
      float mn = fmaxf(rmax[v], lm);
      float al = __expf(rmax[v] - mn);
      rmax[v] = mn;
      float p0 = __expf(sf[0][v] - mn);
      float p1 = __expf(sf[1][v] - mn);
      float psum = p0 + p1;
      for (int msk = 1; msk <= 8; msk <<= 1) psum += __shfl_xor(psum, msk, 32);
      rsum[v] = rsum[v] * al + psum;
      for (int f = 0; f < nacc; ++f) acc[f][v] *= al;
      ps[wid][v + 8 * lh][nn]      = (bf16_t)p0;
      ps[wid][v + 8 * lh][16 + nn] = (bf16_t)p1;
    }
    v16bf paf = frag_row(&ps[wid][nn][0], lh);
    for (int f = 0; f < nacc; ++f) {
      v16bf vb = frag_row(&vsT[f * 16 + nn][0], lh);
      acc[f] = wmma_bf16(paf, vb, acc[f]);
    }
  }
  for (int f = 0; f < nacc; ++f) {
    int dd = f * 16 + nn;
    if (dd < D) {
#pragma unroll
      for (int v = 0; v < 8; ++v) {
        int q = q0 + v + 8 * lh;
        WA[((size_t)b * Nn + q) * HD + h * D + dd] = acc[f][v] / rsum[v];
      }
    }
  }
}

// ================================================================ host
extern "C" void kernel_launch(void* const* d_in, const int* in_sizes, int n_in,
                              void* d_out, int out_size, void* d_ws, size_t ws_size,
                              hipStream_t stream) {
  (void)in_sizes; (void)n_in; (void)out_size; (void)ws_size;
  const float* P[61];
  for (int i = 0; i < 61; ++i) P[i] = (const float*)d_in[i];
  const float* single_act = P[0];
  const float* pair_act   = P[1];
  const float* smask      = P[2];
  const float* pmask      = P[3];

  float* SA = (float*)d_out;          // [1,256,384]
  float* PA = SA + 98304;             // [256,256,128]

  // workspace layout
  float* S0 = (float*)d_ws;
  float* S1 = S0 + 8388608;
  float* S2 = S1 + 8388608;
  float* S3 = S2 + 8388608;
  float* S4 = S3 + 8388608;
  bf16_t* At = (bf16_t*)(S4 + 8388608);
  bf16_t* Bt = At + 8388608;
  float* NBw = (float*)(Bt + 8388608);   // 65536*8
  float* xs  = NBw + 65536 * 8;          // 256*384
  float* sq  = xs + 98304;
  float* sk  = sq + 98304;
  float* sv  = sk + 98304;
  float* swa = sv + 98304;
  float* sg  = swa + 98304;
  float* sh  = sg + 98304;               // 256*1536
  float* oleft  = sh + 393216;           // 256*32
  float* oright = oleft + 8192;

  auto gemm = [&](const float* A, const float* W, const float* bias, float* out,
                  const float* aux, const float* rmask, const float* res,
                  int M, int K, int Nc, float alpha, int act, int st) {
    dim3 g((M + 63) / 64, (Nc + 31) / 32);
    if ((Nc & 31) == 0)
      gemm_kernel<true><<<g, 256, 0, stream>>>(A, W, bias, out, aux, rmask, res,
                                               K, Nc, alpha, act, st, NSEQ);
    else
      gemm_kernel<false><<<g, 256, 0, stream>>>(A, W, bias, out, aux, rmask, res,
                                                K, Nc, alpha, act, st, NSEQ);
  };

  // init working copies of sa / pa
  copy_kernel<<<(24576 + 255) / 256, 256, 0, stream>>>(single_act, SA, 24576);
  copy_kernel<<<(2097152 + 255) / 256, 256, 0, stream>>>(pair_act, PA, 2097152);

  // ---- 1) outer product mean -> pair residual
  ln_kernel<<<256, 128, 0, stream>>>(SA, P[4], xs, CMD, 0, NSEQ);
  gemm(xs, P[5],            P[6],      oleft,  nullptr, smask, nullptr, 256, 384, 32, 1.f, 0, 0);
  gemm(xs, P[5] + 384 * 32, P[6] + 32, oright, nullptr, smask, nullptr, 256, 384, 32, 1.f, 0, 0);
  opm_kernel<<<dim3(1024, 4), 256, 0, stream>>>(oleft, oright, P[7], P[8], smask, PA);

  // ---- 2) row attention with pair bias
  ln_kernel<<<256, 128, 0, stream>>>(SA, P[9], xs, CMD, 0, NSEQ);
  ln_kernel<<<65536, 128, 0, stream>>>(PA, P[10], S0, CZD, 0, NSEQ);
  gemm(S0, P[11], nullptr, NBw, nullptr, nullptr, nullptr, 65536, 128, 8, 1.f, 0, 0);
  gemm(xs, P[12],               nullptr, sq, nullptr, nullptr, nullptr, 256, 384, 384, 1.f / sqrtf(48.f), 0, 0);
  gemm(xs, P[12] + 147456,      nullptr, sk, nullptr, nullptr, nullptr, 256, 384, 384, 1.f, 0, 0);
  gemm(xs, P[12] + 2 * 147456,  nullptr, sv, nullptr, nullptr, nullptr, 256, 384, 384, 1.f, 0, 0);
  attn_kernel<<<dim3(4, 8, 1), 128, 0, stream>>>(sq, sk, sv, NBw, smask, swa, 8, 48, 64, 0);
  gemm(xs, P[13], P[14], sg, swa, nullptr, nullptr, 256, 384, 384, 1.f, 2, 0);
  gemm(sg, P[15], P[16], SA, nullptr, nullptr, SA, 256, 384, 384, 1.f, 0, 0);

  // ---- 3) single transition
  ln_kernel<<<256, 128, 0, stream>>>(SA, P[17], xs, CMD, 0, NSEQ);
  gemm(xs, P[18], P[19], sh, nullptr, nullptr, nullptr, 256, 384, 1536, 1.f, 1, 0);
  gemm(sh, P[20], P[21], SA, nullptr, nullptr, SA, 256, 1536, 384, 1.f, 0, 0);

  // ---- 4/5) triangle multiplications
  auto tri_mult = [&](int pb, bool outgoing) {
    ln_kernel<<<65536, 128, 0, stream>>>(PA, P[pb + 0], S0, CZD, 0, NSEQ);
    gemm(S0, P[pb + 3],         P[pb + 4],       S1, nullptr, nullptr, nullptr, 65536, 128, 128, 1.f, 2, 0);
    gemm(S0, P[pb + 1],         P[pb + 2],       S2, S1, pmask, nullptr, 65536, 128, 128, 1.f, 0, 0);
    gemm(S0, P[pb + 3] + 16384, P[pb + 4] + 128, S1, nullptr, nullptr, nullptr, 65536, 128, 128, 1.f, 2, 0);
    gemm(S0, P[pb + 1] + 16384, P[pb + 2] + 128, S3, S1, pmask, nullptr, 65536, 128, 128, 1.f, 0, 0);
    if (outgoing) {   // O[i,j,c] = sum_k L[i,k,c]*R[j,k,c] : A[c,i,k]=L[i,k,c], B[c,j,k]=R[j,k,c]
      pack_kernel<<<dim3(256, 16, 8), 256, 0, stream>>>(S2, At, 0);
      pack_kernel<<<dim3(256, 16, 8), 256, 0, stream>>>(S3, Bt, 0);
    } else {          // O[i,j,c] = sum_k L[k,j,c]*R[k,i,c] : A[c,i,k]=R[k,i,c], B[c,j,k]=L[k,j,c]
      pack_kernel<<<dim3(256, 16, 8), 256, 0, stream>>>(S3, At, 1);
      pack_kernel<<<dim3(256, 16, 8), 256, 0, stream>>>(S2, Bt, 1);
    }
    trimul_kernel<<<dim3(256, 16), 256, 0, stream>>>(At, Bt, S4);
    ln_kernel<<<65536, 128, 0, stream>>>(S4, P[pb + 5], S1, CZD, 0, NSEQ);
    gemm(S0, P[pb + 8], P[pb + 9], S2, nullptr, nullptr, nullptr, 65536, 128, 128, 1.f, 2, 0);
    gemm(S1, P[pb + 6], P[pb + 7], PA, S2, nullptr, PA, 65536, 128, 128, 1.f, 0, 0);
  };
  tri_mult(22, true);
  tri_mult(32, false);

  // ---- 6/7) triangle attentions
  auto tri_attn = [&](int pb, bool pc) {
    ln_kernel<<<65536, 128, 0, stream>>>(PA, P[pb + 0], S0, CZD, pc ? 1 : 0, NSEQ);
    gemm(S0, P[pb + 1], nullptr, NBw, nullptr, nullptr, nullptr, 65536, 128, 4, 1.f, 0, 0);
    gemm(S0, P[pb + 2],         nullptr, S1, nullptr, nullptr, nullptr, 65536, 128, 128, 1.f / sqrtf(32.f), 0, 0);
    gemm(S0, P[pb + 2] + 16384, nullptr, S2, nullptr, nullptr, nullptr, 65536, 128, 128, 1.f, 0, 0);
    gemm(S0, P[pb + 2] + 32768, nullptr, S3, nullptr, nullptr, nullptr, 65536, 128, 128, 1.f, 0, 0);
    attn_kernel<<<dim3(4, 4, 256), 128, 0, stream>>>(S1, S2, S3, NBw, pmask, S4, 4, 32, 32, pc ? 2 : 1);
    gemm(S0, P[pb + 3], P[pb + 4], S1, S4, nullptr, nullptr, 65536, 128, 128, 1.f, 2, 0);
    gemm(S1, P[pb + 5], P[pb + 6], PA, nullptr, nullptr, PA, 65536, 128, 128, 1.f, 0, pc ? 1 : 0);
  };
  tri_attn(42, false);
  tri_attn(49, true);

  // ---- 8) pair transition (chunked to bound workspace)
  ln_kernel<<<65536, 128, 0, stream>>>(PA, P[56], S0, CZD, 0, NSEQ);
  for (int ch = 0; ch < 4; ++ch) {
    float* Xc = S0 + (size_t)ch * 16384 * 128;
    float* Pc = PA + (size_t)ch * 16384 * 128;
    gemm(Xc, P[57], P[58], S1, nullptr, nullptr, nullptr, 16384, 128, 512, 1.f, 1, 0);
    gemm(S1, P[59], P[60], Pc, nullptr, nullptr, Pc, 16384, 512, 128, 1.f, 0, 0);
  }
}
